// MoE_14164802142243
// MI455X (gfx1250) — matrix operand order repaired and checked by reading, not compiled, not run on hardware
//
#include <hip/hip_runtime.h>
#include <hip/hip_bf16.h>
#include <math.h>
#include <stdint.h>

#define T_TOK 16384
#define D_DIM 1024
#define F_DIM 4096
#define E_NUM 8
#define CAP   2048

#define BM 128
#define BN 128
#define BK 64
#define LDK (BK + 8)   // 72 bf16 = 144B row stride (multiple of 16B -> aligned vector LDS ops)

typedef __bf16 bf16;
typedef __attribute__((ext_vector_type(16))) __bf16 v16bf;
typedef __attribute__((ext_vector_type(8)))  __bf16 bf16x8;
typedef __attribute__((ext_vector_type(8)))  float   v8f;
typedef __attribute__((ext_vector_type(4)))  float   f32x4;

// ---- CDNA5 async global->LDS path (guarded; falls back to sync copy) ----
#if defined(__has_builtin)
#if __has_builtin(__builtin_amdgcn_global_load_async_to_lds_b128)
#define ASYNC_LDS 1
#endif
#if __has_builtin(__builtin_amdgcn_s_wait_asynccnt)
#define HAVE_WAIT_ASYNC 1
#endif
#endif
#ifndef ASYNC_LDS
#define ASYNC_LDS 0
#endif

// builtin prototype (from hipcc diagnostic): param0 = 'int __vector(4) __device__ *'
typedef int v4i __attribute__((__vector_size__(4 * sizeof(int))));
typedef __attribute__((address_space(1))) v4i* gv4i_p;
typedef __attribute__((address_space(3))) v4i* lv4i_p;

__device__ __forceinline__ void copy16(const bf16* g, bf16* l) {
#if ASYNC_LDS
  // Low 32 bits of a generic LDS address are the DS byte address (ISA 10.2),
  // so build the AS(3) pointer via inttoptr; AS(1) numeric value == generic.
  __builtin_amdgcn_global_load_async_to_lds_b128(
      (gv4i_p)(uintptr_t)g, (lv4i_p)(uint32_t)(uintptr_t)l, 0, 0);
#else
  *(bf16x8*)l = *(const bf16x8*)g;
#endif
}

__device__ __forceinline__ void copy_fence() {
#if ASYNC_LDS
#if defined(HAVE_WAIT_ASYNC)
  __builtin_amdgcn_s_wait_asynccnt(0);
#else
  asm volatile("s_wait_asynccnt 0x0" ::: "memory");
#endif
#endif
}

// ---------------- Router: one wave32 per token ----------------
__global__ __launch_bounds__(256) void router_kernel(
    const float* __restrict__ x, const float* __restrict__ Wr,
    int* __restrict__ eidx, float* __restrict__ ptok) {
  int tok  = (blockIdx.x * 256 + threadIdx.x) >> 5;
  int lane = threadIdx.x & 31;
  const float* xr = x + (size_t)tok * D_DIM;
  float acc[E_NUM];
#pragma unroll
  for (int e = 0; e < E_NUM; ++e) acc[e] = 0.f;
  for (int k = lane; k < D_DIM; k += 32) {
    float xv = xr[k];
    const float* wr = Wr + k * E_NUM;
#pragma unroll
    for (int e = 0; e < E_NUM; ++e) acc[e] = fmaf(xv, wr[e], acc[e]);
  }
#pragma unroll
  for (int off = 16; off > 0; off >>= 1) {
#pragma unroll
    for (int e = 0; e < E_NUM; ++e) acc[e] += __shfl_xor(acc[e], off, 32);
  }
  if (lane == 0) {
    float m = acc[0]; int ai = 0;
#pragma unroll
    for (int e = 1; e < E_NUM; ++e) { if (acc[e] > m) { m = acc[e]; ai = e; } }
    float s = 0.f;
#pragma unroll
    for (int e = 0; e < E_NUM; ++e) s += __expf(acc[e] - m);
    eidx[tok] = ai;
    ptok[tok] = 1.0f / s;   // softmax prob of the argmax expert
  }
}

// ---------------- Dispatch: single block, deterministic token-order ranking ----------------
__global__ __launch_bounds__(256) void dispatch_kernel(
    const int* __restrict__ eidx, int* __restrict__ dispatch) {
  __shared__ int sh[256][E_NUM];
  int tid = threadIdx.x;
  const int chunk = T_TOK / 256;             // 64 tokens per thread
  int t0 = tid * chunk;
  int loc[E_NUM];
#pragma unroll
  for (int e = 0; e < E_NUM; ++e) loc[e] = 0;
  for (int i = 0; i < chunk; ++i) loc[eidx[t0 + i]]++;
#pragma unroll
  for (int e = 0; e < E_NUM; ++e) sh[tid][e] = loc[e];
  for (int i = tid; i < E_NUM * CAP; i += 256) dispatch[i] = -1;
  __syncthreads();
  int base[E_NUM];
#pragma unroll
  for (int e = 0; e < E_NUM; ++e) {
    int b = 0;
    for (int j = 0; j < tid; ++j) b += sh[j][e];
    base[e] = b;
  }
  for (int i = 0; i < chunk; ++i) {
    int t = t0 + i;
    int e = eidx[t];
    int r = base[e]++;
    if (r < CAP) dispatch[e * CAP + r] = t;
  }
}

// ---------------- fp32 -> bf16 elementwise (x) ----------------
__global__ __launch_bounds__(256) void cvt_x_kernel(
    const float* __restrict__ in, bf16* __restrict__ out) {
  size_t i = ((size_t)blockIdx.x * 256 + threadIdx.x) * 8;
  f32x4 a = *(const f32x4*)(in + i);
  f32x4 b = *(const f32x4*)(in + i + 4);
  bf16x8 o;
  o[0] = (bf16)a[0]; o[1] = (bf16)a[1]; o[2] = (bf16)a[2]; o[3] = (bf16)a[3];
  o[4] = (bf16)b[0]; o[5] = (bf16)b[1]; o[6] = (bf16)b[2]; o[7] = (bf16)b[3];
  *(bf16x8*)(out + i) = o;
}

// ------- fp32 [E][K][N] -> bf16 [E][N][K] tiled transpose-convert (weights) -------
__global__ __launch_bounds__(256) void cvt_tr_kernel(
    const float* __restrict__ in, bf16* __restrict__ out, int K, int N) {
  __shared__ __align__(16) bf16 tile[64][72];
  int ktiles = K >> 6, ntiles = N >> 6;
  int bx  = blockIdx.x;
  int e   = bx / (ktiles * ntiles);
  int rem = bx % (ktiles * ntiles);
  int k0  = (rem / ntiles) << 6;
  int n0  = (rem % ntiles) << 6;
  const float* ine  = in  + (size_t)e * K * N;
  bf16*        oute = out + (size_t)e * K * N;
  int tid = threadIdx.x;
#pragma unroll
  for (int p = 0; p < 4; ++p) {
    int idx = p * 256 + tid;        // 64 k-rows x 16 float4
    int r   = idx >> 4;
    int c4  = (idx & 15) << 2;
    f32x4 v = *(const f32x4*)(ine + (size_t)(k0 + r) * N + n0 + c4);
    tile[c4 + 0][r] = (bf16)v[0];
    tile[c4 + 1][r] = (bf16)v[1];
    tile[c4 + 2][r] = (bf16)v[2];
    tile[c4 + 3][r] = (bf16)v[3];
  }
  __syncthreads();
#pragma unroll
  for (int p = 0; p < 2; ++p) {
    int idx = p * 256 + tid;        // 64 n-rows x 8 chunks
    int n   = idx >> 3;
    int c8  = (idx & 7) << 3;
    *(bf16x8*)(oute + (size_t)(n0 + n) * K + k0 + c8) = *(const bf16x8*)&tile[n][c8];
  }
}

// WMMA fragment loads from LDS (layouts per CDNA5 ISA 7.12.2)
__device__ __forceinline__ v16bf load_a_frag(const bf16* pa) {
  bf16x8 lo = *(const bf16x8*)pa;          // K = kk + khalf + 0..7
  bf16x8 hi = *(const bf16x8*)(pa + 16);   // K = kk + 16 + khalf + 0..7
  return __builtin_shufflevector(lo, hi, 0,1,2,3,4,5,6,7,8,9,10,11,12,13,14,15);
}
__device__ __forceinline__ v16bf load_b_frag(const bf16* pb) {
  bf16x8 lo = *(const bf16x8*)pb;          // K = kk + khalf + 0..7
  bf16x8 hi = *(const bf16x8*)(pb + 16);   // K = kk + 16 + khalf + 0..7
  return __builtin_shufflevector(lo, hi, 0,1,2,3,4,5,6,7,8,9,10,11,12,13,14,15);
}

// ---------------- GEMM1: H = relu(gather(xb) @ W1t[e]^T), bf16 out ----------------
__global__ __launch_bounds__(256) void gemm1_kernel(
    const bf16* __restrict__ xb, const bf16* __restrict__ W1t,
    const int* __restrict__ dispatch, bf16* __restrict__ Hb) {
  __shared__ __align__(16) bf16 As[2][BM][LDK];   // [m][k]
  __shared__ __align__(16) bf16 Bs[2][BN][LDK];   // [n][k]
  __shared__ int tokRow[BM];

  const int nTiles = F_DIM / BN;               // 32
  const int mTiles = CAP / BM;                 // 16
  int bx  = blockIdx.x;
  int e   = bx / (mTiles * nTiles);
  int rem = bx % (mTiles * nTiles);
  int m0  = (rem / nTiles) * BM;
  int n0  = (rem % nTiles) * BN;

  int tid  = threadIdx.x;
  int lane = tid & 31;
  int wave = tid >> 5;
  int wm   = (wave & 3) * 32;
  int wn   = (wave >> 2) * 64;

  if (tid < BM) tokRow[tid] = dispatch[e * CAP + m0 + tid];
  __syncthreads();

  // per-thread fixed gather rows (A tile): row = (p*256+tid)>>3
  int tokA[4];
#pragma unroll
  for (int p = 0; p < 4; ++p) {
    int t = tokRow[(p * 256 + tid) >> 3];
    tokA[p] = t < 0 ? 0 : t;     // dropped slots gather row 0; result discarded later
  }

  const bf16* W1e = W1t + (size_t)e * F_DIM * D_DIM;   // [F][D]

  auto issueA = [&](int k0, int b) {
#pragma unroll
    for (int p = 0; p < 4; ++p) {
      int idx = p * 256 + tid;
      int row = idx >> 3;
      int c8  = (idx & 7) << 3;
      copy16(xb + (size_t)tokA[p] * D_DIM + k0 + c8, &As[b][row][c8]);
    }
  };
  auto issueB = [&](int k0, int b) {
#pragma unroll
    for (int p = 0; p < 4; ++p) {
      int idx = p * 256 + tid;
      int n   = idx >> 3;
      int c8  = (idx & 7) << 3;
      copy16(W1e + (size_t)(n0 + n) * D_DIM + k0 + c8, &Bs[b][n][c8]);
    }
  };

  v8f acc[2][4];
#pragma unroll
  for (int mi = 0; mi < 2; ++mi)
#pragma unroll
    for (int ni = 0; ni < 4; ++ni) acc[mi][ni] = (v8f)0.0f;

  const int NIT = D_DIM / BK;   // 16
  issueA(0, 0); issueB(0, 0);
  int buf = 0;
  for (int it = 0; it < NIT; ++it) {
    copy_fence();
    __syncthreads();
    if (it + 1 < NIT) { issueA((it + 1) * BK, buf ^ 1); issueB((it + 1) * BK, buf ^ 1); }

    int khalf = (lane >> 4) << 3;
    int arow  = wm + (lane & 15);
#pragma unroll
    for (int kk = 0; kk < BK; kk += 32) {
      v16bf a[2], b[4];
#pragma unroll
      for (int mi = 0; mi < 2; ++mi)
        a[mi] = load_a_frag(&As[buf][arow + mi * 16][kk + khalf]);
      int kstart = kk + ((lane >> 4) << 3);
#pragma unroll
      for (int ni = 0; ni < 4; ++ni)
        b[ni] = load_b_frag(&Bs[buf][wn + ni * 16 + (lane & 15)][kstart]);
#pragma unroll
      for (int mi = 0; mi < 2; ++mi)
#pragma unroll
        for (int ni = 0; ni < 4; ++ni)
          acc[mi][ni] = __builtin_amdgcn_wmma_f32_16x16x32_bf16(
              false, a[mi], false, b[ni], (short)0, acc[mi][ni], false, false);
    }
    buf ^= 1;
  }

  bf16* He = Hb + (size_t)e * CAP * F_DIM;
#pragma unroll
  for (int mi = 0; mi < 2; ++mi) {
#pragma unroll
    for (int ni = 0; ni < 4; ++ni) {
      int mbase = m0 + wm + mi * 16 + ((lane >> 4) << 3);
      int n     = n0 + wn + ni * 16 + (lane & 15);
#pragma unroll
      for (int r = 0; r < 8; ++r) {
        float v = acc[mi][ni][r];
        v = v > 0.f ? v : 0.f;
        He[(size_t)(mbase + r) * F_DIM + n] = (bf16)v;
      }
    }
  }
}

// ---------------- GEMM2: y[token] = p_tok * (H_e @ W2t[e]^T) ----------------
__global__ __launch_bounds__(256) void gemm2_kernel(
    const bf16* __restrict__ Hb, const bf16* __restrict__ W2t,
    const int* __restrict__ dispatch, const float* __restrict__ ptok,
    float* __restrict__ y) {
  __shared__ __align__(16) bf16 As[2][BM][LDK];
  __shared__ __align__(16) bf16 Bs[2][BN][LDK];
  __shared__ int   tokRow[BM];
  __shared__ float pRow[BM];

  const int nTiles = D_DIM / BN;               // 8
  const int mTiles = CAP / BM;                 // 16
  int bx  = blockIdx.x;
  int e   = bx / (mTiles * nTiles);
  int rem = bx % (mTiles * nTiles);
  int m0  = (rem / nTiles) * BM;
  int n0  = (rem % nTiles) * BN;

  int tid  = threadIdx.x;
  int lane = tid & 31;
  int wave = tid >> 5;
  int wm   = (wave & 3) * 32;
  int wn   = (wave >> 2) * 64;

  if (tid < BM) {
    int t = dispatch[e * CAP + m0 + tid];
    tokRow[tid] = t;
    pRow[tid]   = (t >= 0) ? ptok[t] : 0.f;
  }

  const bf16* He  = Hb  + (size_t)e * CAP * F_DIM;     // [CAP][F]
  const bf16* W2e = W2t + (size_t)e * D_DIM * F_DIM;   // [D][F]

  auto issueA = [&](int k0, int b) {
#pragma unroll
    for (int p = 0; p < 4; ++p) {
      int idx = p * 256 + tid;
      int row = idx >> 3;
      int c8  = (idx & 7) << 3;
      copy16(He + (size_t)(m0 + row) * F_DIM + k0 + c8, &As[b][row][c8]);
    }
  };
  auto issueB = [&](int k0, int b) {
#pragma unroll
    for (int p = 0; p < 4; ++p) {
      int idx = p * 256 + tid;
      int n   = idx >> 3;
      int c8  = (idx & 7) << 3;
      copy16(W2e + (size_t)(n0 + n) * F_DIM + k0 + c8, &Bs[b][n][c8]);
    }
  };

  v8f acc[2][4];
#pragma unroll
  for (int mi = 0; mi < 2; ++mi)
#pragma unroll
    for (int ni = 0; ni < 4; ++ni) acc[mi][ni] = (v8f)0.0f;

  const int NIT = F_DIM / BK;   // 64
  issueA(0, 0); issueB(0, 0);
  int buf = 0;
  for (int it = 0; it < NIT; ++it) {
    copy_fence();
    __syncthreads();
    if (it + 1 < NIT) { issueA((it + 1) * BK, buf ^ 1); issueB((it + 1) * BK, buf ^ 1); }

    int khalf = (lane >> 4) << 3;
    int arow  = wm + (lane & 15);
#pragma unroll
    for (int kk = 0; kk < BK; kk += 32) {
      v16bf a[2], b[4];
#pragma unroll
      for (int mi = 0; mi < 2; ++mi)
        a[mi] = load_a_frag(&As[buf][arow + mi * 16][kk + khalf]);
      int kstart = kk + ((lane >> 4) << 3);
#pragma unroll
      for (int ni = 0; ni < 4; ++ni)
        b[ni] = load_b_frag(&Bs[buf][wn + ni * 16 + (lane & 15)][kstart]);
#pragma unroll
      for (int mi = 0; mi < 2; ++mi)
#pragma unroll
        for (int ni = 0; ni < 4; ++ni)
          acc[mi][ni] = __builtin_amdgcn_wmma_f32_16x16x32_bf16(
              false, a[mi], false, b[ni], (short)0, acc[mi][ni], false, false);
    }
    buf ^= 1;
  }

#pragma unroll
  for (int mi = 0; mi < 2; ++mi) {
#pragma unroll
    for (int ni = 0; ni < 4; ++ni) {
      int mb = wm + mi * 16 + ((lane >> 4) << 3);
      int n  = n0 + wn + ni * 16 + (lane & 15);
#pragma unroll
      for (int r = 0; r < 8; ++r) {
        int ml  = mb + r;
        int tok = tokRow[ml];
        if (tok >= 0) y[(size_t)tok * D_DIM + n] = pRow[ml] * acc[mi][ni][r];
      }
    }
  }
}

extern "C" void kernel_launch(void* const* d_in, const int* in_sizes, int n_in,
                              void* d_out, int out_size, void* d_ws, size_t ws_size,
                              hipStream_t stream) {
  const float* x  = (const float*)d_in[0];
  const float* Wr = (const float*)d_in[1];
  const float* W1 = (const float*)d_in[2];
  const float* W2 = (const float*)d_in[3];
  float* y = (float*)d_out;

  char* ws = (char*)d_ws;
  const size_t MB = 1024 * 1024;
  int*   eidx     = (int*)(ws + 0);                    // 64 KB
  float* ptokp    = (float*)(ws + (64 << 10));         // 64 KB
  int*   dispatch = (int*)(ws + (128 << 10));          // 64 KB
  bf16*  xb       = (bf16*)(ws + (256 << 10));         // 32 MB  [T][D]
  bf16*  W1t      = (bf16*)(ws + (256 << 10) + 32 * MB);   // 64 MB  [E][F][D]
  bf16*  W2t      = (bf16*)(ws + (256 << 10) + 96 * MB);   // 64 MB  [E][D][F]
  bf16*  Hb       = (bf16*)(ws + (256 << 10) + 160 * MB);  // 128 MB [E][CAP][F]

  (void)hipMemsetAsync(y, 0, (size_t)T_TOK * D_DIM * sizeof(float), stream);
  router_kernel<<<T_TOK / 8, 256, 0, stream>>>(x, Wr, eidx, ptokp);
  dispatch_kernel<<<1, 256, 0, stream>>>(eidx, dispatch);
  cvt_x_kernel<<<(T_TOK * D_DIM) / (256 * 8), 256, 0, stream>>>(x, xb);
  cvt_tr_kernel<<<E_NUM * (D_DIM / 64) * (F_DIM / 64), 256, 0, stream>>>(W1, W1t, D_DIM, F_DIM);
  cvt_tr_kernel<<<E_NUM * (F_DIM / 64) * (D_DIM / 64), 256, 0, stream>>>(W2, W2t, F_DIM, D_DIM);
  gemm1_kernel<<<E_NUM * (CAP / BM) * (F_DIM / BN), 256, 0, stream>>>(xb, W1t, dispatch, Hb);
  gemm2_kernel<<<E_NUM * (CAP / BM) * (D_DIM / BN), 256, 0, stream>>>(Hb, W2t, dispatch, ptokp, y);
}
